// LocalRouter_87892210745384
// MI455X (gfx1250) — matrix operand order repaired and checked by reading, not compiled, not run on hardware
//
#include <hip/hip_runtime.h>
#include <hip/hip_bf16.h>

// ---------------------------------------------------------------------------
// CDNA5 (gfx1250): all GEMM-like work through v_wmma_f32_16x16x32_bf16
// (wave32), GEMM tile staging through the Tensor Data Mover when available.
// ---------------------------------------------------------------------------

typedef __attribute__((ext_vector_type(16))) __bf16 v16bf;
typedef __attribute__((ext_vector_type(8)))  float  v8f;

#if defined(__HIP_DEVICE_COMPILE__) && \
    __has_builtin(__builtin_amdgcn_tensor_load_to_lds) && \
    __has_builtin(__builtin_amdgcn_s_wait_tensorcnt)
#define USE_TDM 1
typedef __attribute__((ext_vector_type(4))) unsigned int v4u;
typedef __attribute__((ext_vector_type(8))) int          v8i;
typedef __attribute__((ext_vector_type(4))) int          v4i;
#else
#define USE_TDM 0
#endif

__device__ __forceinline__ v8f vzero8() {
    v8f z;
#pragma unroll
    for (int i = 0; i < 8; ++i) z[i] = 0.0f;
    return z;
}

// float -> bf16, round-to-nearest-even
__device__ __forceinline__ unsigned short f2bf_rn(float f) {
    unsigned int u = __float_as_uint(f);
    unsigned int r = u + 0x7fffu + ((u >> 16) & 1u);
    return (unsigned short)(r >> 16);
}
__device__ __forceinline__ unsigned int pack2bf(float a, float b) {
    return (unsigned int)f2bf_rn(a) | ((unsigned int)f2bf_rn(b) << 16);
}

// 16-half fragment from two 8-half (16B) groups (2x b128 loads)
__device__ __forceinline__ v16bf ldfrag2(const unsigned short* p0,
                                         const unsigned short* p1) {
    union { v16bf v; uint4 q[2]; } u;
    u.q[0] = *(const uint4*)p0;
    u.q[1] = *(const uint4*)p1;
    return u.v;
}

#if USE_TDM
// Issue a TDM load of a [rows=128 x 32 halves] tile (row stride = K halves)
// from global bf16 into LDS.  D# layout per CDNA5 ISA.
// 6-arg builtin form: (g0 u32x4, g1 i32x8, g2 i32x4, g3 i32x4, i32x8, cpol).
__device__ __forceinline__ void tdm_load_tile(const unsigned short* gbase,
                                              unsigned int lds_off, int K) {
    const unsigned long long ga = (unsigned long long)(size_t)gbase;
    v4u g0;
    g0[0] = 1u;                                  // count=1, user mode
    g0[1] = lds_off;                             // lds_addr
    g0[2] = (unsigned int)ga;                    // global_addr[31:0]
    g0[3] = (unsigned int)((ga >> 32) & 0x1FFFFFFull) | (2u << 30);  // type=2
    const unsigned int td1 = 1u << 20;           // generous tensor_dim1
    v8i g1;
    g1[0] = (int)(1u << 16);                     // wg_mask=0, data_size=2B
    g1[1] = (int)(((unsigned)K & 0xFFFFu) << 16);            // dim0 lo16
    g1[2] = (int)(((unsigned)K >> 16) | ((td1 & 0xFFFFu) << 16));
    g1[3] = (int)((td1 >> 16) | (32u << 16));    // dim1 hi | tile_dim0=32
    g1[4] = 128;                                 // tile_dim1=128, tile_dim2=0
    g1[5] = K;                                   // tensor_dim0_stride lo32
    g1[6] = 0;                                   // stride hi, dim1_stride lo
    g1[7] = 0;
    const v4i zz4 = {0, 0, 0, 0};
    const v8i zz8 = {0, 0, 0, 0, 0, 0, 0, 0};
    __builtin_amdgcn_tensor_load_to_lds(g0, g1, zz4, zz4, zz8, 0);
}
#endif

// ---------------------------------------------------------------------------
// f32 -> bf16 bulk convert (4 elements / thread)
// ---------------------------------------------------------------------------
__global__ __launch_bounds__(256)
void cvt_bf16_kernel(const float* __restrict__ src,
                     unsigned short* __restrict__ dst, long long n4) {
    const long long i = (long long)blockIdx.x * 256 + threadIdx.x;
    if (i < n4) {
        const float4 v = ((const float4*)src)[i];
        ((uint2*)dst)[i] = make_uint2(pack2bf(v.x, v.y), pack2bf(v.z, v.w));
    }
}

// ---------------------------------------------------------------------------
// Transpose-convert weights: Wt[c, k] = bf16(W[k, c]), 32x32 LDS tiles.
// ---------------------------------------------------------------------------
__global__ __launch_bounds__(256)
void transpose_bf16_kernel(const float* __restrict__ W,
                           unsigned short* __restrict__ Wt,
                           int Kd, int Nout, int ldw) {
    __shared__ float t[32][33];
    const int tid = threadIdx.x;
    const int c0 = blockIdx.x * 32;
    const int k0 = blockIdx.y * 32;
#pragma unroll
    for (int i = 0; i < 4; ++i) {
        const int id = tid + 256 * i;
        const int kk = id >> 5, cc = id & 31;
        t[kk][cc] = W[(size_t)(k0 + kk) * ldw + c0 + cc];
    }
    __syncthreads();
#pragma unroll
    for (int i = 0; i < 4; ++i) {
        const int id = tid + 256 * i;
        const int cc = id >> 5, kk = id & 31;
        Wt[(size_t)(c0 + cc) * Kd + k0 + kk] = f2bf_rn(t[kk][cc]);
    }
}

// ---------------------------------------------------------------------------
// Tiled GEMM: C[M,Nout] = Ab[M,K](bf16) @ Wt[Nout,K](bf16,transposed) (+bias)
// Block 128x128xK32, 256 thr = 8 waves, wave = 64x32 (4x2 WMMA accums).
// Staging via TDM (double-buffered) when available, else b128 copies.
// ---------------------------------------------------------------------------
template <bool OUT_BF16, bool HAS_BIAS, bool ACCUM>
__global__ __launch_bounds__(256)
void gemm_wmma_bf16(const unsigned short* __restrict__ Ab,
                    const unsigned short* __restrict__ Wt,
                    const float* __restrict__ bias,
                    void* __restrict__ Cp,
                    int M, int K, int Nout) {
    __shared__ alignas(16) unsigned short As[2][128 * 32];  // [row][k]
    __shared__ alignas(16) unsigned short Bs[2][128 * 32];  // [col][k]

    const int tid   = threadIdx.x;
    const int lane  = tid & 31;
    const int wave  = tid >> 5;
    const int wr    = wave >> 2;   // 0..1
    const int wc    = wave & 3;    // 0..3
    const int l16   = lane & 15;
    const int lhalf = (lane < 16) ? 0 : 1;
    const int rowBlock = blockIdx.y * 128;
    const int colBlock = blockIdx.x * 128;

    v8f acc[4][2];
#pragma unroll
    for (int i = 0; i < 4; ++i)
#pragma unroll
        for (int j = 0; j < 2; ++j) acc[i][j] = vzero8();

    const int nk = K >> 5;

    const unsigned short* aBase = Ab + (size_t)rowBlock * K;
    const unsigned short* bBase = Wt + (size_t)colBlock * K;

#if !USE_TDM
    auto stage = [&](int kt, int buf) {
        const int k0 = kt << 5;
#pragma unroll
        for (int i = 0; i < 2; ++i) {
            const int id = tid + 256 * i;          // 512 uint4 per tile
            const int r = id >> 2, o = id & 3;
            *(uint4*)&As[buf][r * 32 + o * 8] =
                *(const uint4*)(aBase + (size_t)r * K + k0 + o * 8);
        }
#pragma unroll
        for (int i = 0; i < 2; ++i) {
            const int id = tid + 256 * i;
            const int r = id >> 2, o = id & 3;
            *(uint4*)&Bs[buf][r * 32 + o * 8] =
                *(const uint4*)(bBase + (size_t)r * K + k0 + o * 8);
        }
    };
#endif

#if USE_TDM
    if (wave == 0) {
        tdm_load_tile(aBase, (unsigned int)(size_t)&As[0][0], K);
        tdm_load_tile(bBase, (unsigned int)(size_t)&Bs[0][0], K);
        __builtin_amdgcn_s_wait_tensorcnt(0);
    }
#else
    stage(0, 0);
#endif
    __syncthreads();

    for (int kt = 0; kt < nk; ++kt) {
        const int cur = kt & 1;
#if USE_TDM
        if (kt + 1 < nk && wave == 0) {
            const int k1 = (kt + 1) << 5;
            tdm_load_tile(aBase + k1, (unsigned int)(size_t)&As[cur ^ 1][0], K);
            tdm_load_tile(bBase + k1, (unsigned int)(size_t)&Bs[cur ^ 1][0], K);
        }
#else
        if (kt + 1 < nk) stage(kt + 1, cur ^ 1);
#endif

        v16bf af[4], bf[2];
#pragma unroll
        for (int i = 0; i < 4; ++i) {
            // A frag: lane<16 -> K {0..7,16..23}; lane>=16 -> {8..15,24..31}
            const unsigned short* p =
                &As[cur][(wr * 64 + i * 16 + l16) * 32 + lhalf * 8];
            af[i] = ldfrag2(p, p + 16);
        }
#pragma unroll
        for (int j = 0; j < 2; ++j) {
            // B frag: lane<16 -> K {0..7,16..23}; lane>=16 -> {8..15,24..31}
            const unsigned short* p =
                &Bs[cur][(wc * 32 + j * 16 + l16) * 32 + lhalf * 8];
            bf[j] = ldfrag2(p, p + 16);
        }
#pragma unroll
        for (int i = 0; i < 4; ++i)
#pragma unroll
            for (int j = 0; j < 2; ++j)
                acc[i][j] = __builtin_amdgcn_wmma_f32_16x16x32_bf16(
                    false, af[i], false, bf[j], (short)0, acc[i][j], false,
                    false);

#if USE_TDM
        if (wave == 0 && kt + 1 < nk) __builtin_amdgcn_s_wait_tensorcnt(0);
#endif
        __syncthreads();
    }

    // Epilogue: VGPR e = row (e + 8*lhalf), col = lane%16
#pragma unroll
    for (int j = 0; j < 2; ++j) {
        const int c = colBlock + wc * 32 + j * 16 + l16;
        const float bvj = HAS_BIAS ? bias[c] : 0.0f;
#pragma unroll
        for (int i = 0; i < 4; ++i) {
            const int rbase = rowBlock + wr * 64 + i * 16 + lhalf * 8;
#pragma unroll
            for (int e = 0; e < 8; ++e) {
                const size_t off = (size_t)(rbase + e) * Nout + c;
                float val = acc[i][j][e] + bvj;
                if constexpr (OUT_BF16) {
                    ((unsigned short*)Cp)[off] = f2bf_rn(val);
                } else {
                    float* Cf = (float*)Cp;
                    if constexpr (ACCUM) val += Cf[off];
                    Cf[off] = val;
                }
            }
        }
    }
}

// ---------------------------------------------------------------------------
// s[b,n,:] = 0.25 * sum_{w=1..4} silu(A[b,n,:] + Bm[b,n-w,:]), bf16 output.
// ---------------------------------------------------------------------------
__global__ __launch_bounds__(256)
void silu_window_mean(const float* __restrict__ A,
                      const float* __restrict__ Bm,
                      unsigned short* __restrict__ S, int N, int D) {
    const long long idx = (long long)blockIdx.x * 256 + threadIdx.x;
    const int n = (int)((idx / D) % N);
    const float a = A[idx];
    float s = 0.0f;
#pragma unroll
    for (int w = 1; w <= 4; ++w) {
        float pre = a + ((n >= w) ? Bm[idx - (long long)w * D] : 0.0f);
        s += pre / (1.0f + __expf(-pre));  // silu
    }
    S[idx] = f2bf_rn(0.25f * s);
}

// ---------------------------------------------------------------------------
// Fused causal top-8 attention. 1 WG (128 thr = 4 waves) per 16 query rows.
// Scores via WMMA bf16, streaming per-row top-8, softmax over 8, 8-way
// weighted gather of v.  glob written as bf16 (feeds the final GEMM).
// ---------------------------------------------------------------------------
__global__ __launch_bounds__(128)
void attn_topk_kernel(const unsigned short* __restrict__ qb,
                      const unsigned short* __restrict__ kb,
                      const float* __restrict__ v,
                      unsigned short* __restrict__ glob,
                      int N, int D, float scale) {
    __shared__ alignas(16) unsigned short qt[16 * 1024];
    __shared__ alignas(16) float sc[16][256];
    __shared__ float wts[16][8];
    __shared__ int   idxs[16][8];

    const int tilesPerB = N >> 4;
    const int b    = blockIdx.x / tilesPerB;
    const int tile = blockIdx.x % tilesPerB;
    const int r0   = tile << 4;
    const int tid  = threadIdx.x;
    const int lane = tid & 31;
    const int wave = tid >> 5;
    const int l16  = lane & 15;
    const int lhalf = (lane < 16) ? 0 : 1;
    const size_t bN = (size_t)b * N;

    {   // stage q tile (16 x D bf16)
        const int upr = D >> 3;
        for (int id = tid; id < 16 * upr; id += 128) {
            const int r = id / upr, o = id % upr;
            ((uint4*)qt)[(size_t)r * upr + o] =
                *(const uint4*)(qb + (bN + r0 + r) * D + o * 8);
        }
    }

    float tv[8];
    int   ti[8];
#pragma unroll
    for (int q = 0; q < 8; ++q) { tv[q] = -1e30f; ti[q] = 0; }
    float mv = -1e30f;
    int   am = 0;

    const int mMax    = r0 + 15;
    const int nChunks = (mMax + 256) >> 8;

    __syncthreads();

    for (int ch = 0; ch < nChunks; ++ch) {
        const int c0 = ch << 8;
        for (int u = 0; u < 4; ++u) {
            const int tt = wave + 4 * u;
            const int m0 = c0 + tt * 16;
            v8f accv = vzero8();
            if (m0 <= mMax) {
                const unsigned short* kbase = kb + (bN + m0 + l16) * D;
                for (int k0 = 0; k0 < D; k0 += 32) {
                    const unsigned short* ap = &qt[l16 * D + k0 + lhalf * 8];
                    const v16bf afr = ldfrag2(ap, ap + 16);
                    const unsigned short* bp = kbase + k0 + lhalf * 8;
                    const v16bf bfr = ldfrag2(bp, bp + 16);
                    accv = __builtin_amdgcn_wmma_f32_16x16x32_bf16(
                        false, afr, false, bfr, (short)0, accv, false, false);
                }
            }
#pragma unroll
            for (int e = 0; e < 8; ++e) {
                const int rl = e + lhalf * 8;
                const int mg = m0 + l16;
                sc[rl][tt * 16 + l16] =
                    (mg <= r0 + rl) ? accv[e] * scale : -1e30f;
            }
        }
        __syncthreads();
        if (tid < 16) {
            for (int j = 0; j < 256; ++j) {
                const float val = sc[tid][j];
                if (val > mv) {
                    tv[am] = val;
                    ti[am] = c0 + j;
                    mv = tv[0]; am = 0;
#pragma unroll
                    for (int q = 1; q < 8; ++q)
                        if (tv[q] < mv) { mv = tv[q]; am = q; }
                }
            }
        }
        __syncthreads();
    }

    if (tid < 16) {
        float mx = tv[0];
#pragma unroll
        for (int q = 1; q < 8; ++q) mx = fmaxf(mx, tv[q]);
        float w[8], s = 0.0f;
#pragma unroll
        for (int q = 0; q < 8; ++q) { w[q] = __expf(tv[q] - mx); s += w[q]; }
        const float inv = 1.0f / s;
#pragma unroll
        for (int q = 0; q < 8; ++q) {
            wts[tid][q]  = w[q] * inv;
            idxs[tid][q] = ti[q];
        }
    }
    __syncthreads();

    {   // glob[row] = sum_j w_j * v[idx_j]   (bf16 out)
        const int row = tid >> 3;
        const int seg = tid & 7;
        float        wr8[8];
        const float* vp[8];
#pragma unroll
        for (int q = 0; q < 8; ++q) {
            wr8[q] = wts[row][q];
            vp[q]  = v + (bN + idxs[row][q]) * D;
        }
        unsigned short* gout = glob + (bN + r0 + row) * D;
        const int cbase = seg * (D >> 3);
        for (int g = 0; g < (D >> 3); g += 4) {
            const int col = cbase + g;
            float a0 = 0.f, a1 = 0.f, a2 = 0.f, a3 = 0.f;
#pragma unroll
            for (int q = 0; q < 8; ++q) {
                const float4 vv = *(const float4*)(vp[q] + col);
                a0 = fmaf(wr8[q], vv.x, a0);
                a1 = fmaf(wr8[q], vv.y, a1);
                a2 = fmaf(wr8[q], vv.z, a2);
                a3 = fmaf(wr8[q], vv.w, a3);
            }
            *(uint2*)(gout + col) =
                make_uint2(pack2bf(a0, a1), pack2bf(a2, a3));
        }
    }
}

// ---------------------------------------------------------------------------
extern "C" void kernel_launch(void* const* d_in, const int* in_sizes, int n_in,
                              void* d_out, int out_size, void* d_ws,
                              size_t ws_size, hipStream_t stream) {
    const float* mu  = (const float*)d_in[0];
    const float* Wq  = (const float*)d_in[1];
    const float* bq  = (const float*)d_in[2];
    const float* Wk  = (const float*)d_in[3];
    const float* bk  = (const float*)d_in[4];
    const float* Wv  = (const float*)d_in[5];
    const float* bv  = (const float*)d_in[6];
    const float* Wm1 = (const float*)d_in[7];
    const float* bm1 = (const float*)d_in[8];
    const float* Wm2 = (const float*)d_in[9];
    const float* bm2 = (const float*)d_in[10];
    const float* Wo  = (const float*)d_in[11];
    const float* bo  = (const float*)d_in[12];

    const int D = in_sizes[2];             // 1024
    const int M = in_sizes[0] / D;         // B*N = 8192
    const int N = (M >= 2048) ? 2048 : M;  // reference N
    const float scale = 1.0f / sqrtf((float)D);

    // Workspace layout (bytes)
    char* w = (char*)d_ws;
    const size_t szBf = (size_t)M * D * sizeof(unsigned short);
    const size_t szF  = (size_t)M * D * sizeof(float);
    unsigned short* mub  = (unsigned short*)(w);            // later: s, glob
    unsigned short* qb   = (unsigned short*)(w + szBf);
    unsigned short* kbuf = (unsigned short*)(w + 2 * szBf);
    float*          vbuf = (float*)(w + 3 * szBf);
    float*          Abuf = (float*)(w + 3 * szBf + szF);    // later: local(bf16)
    float*          Bbuf = (float*)(w + 3 * szBf + 2 * szF);
    unsigned short* Wt   = (unsigned short*)(w + 3 * szBf + 3 * szF);
    unsigned short* sb     = mub;                        // s over mu-bf16
    unsigned short* localb = (unsigned short*)Abuf;      // local over A
    unsigned short* globb  = mub;                        // glob over s (dead)
    (void)ws_size; (void)n_in; (void)out_size;

    const dim3 blk(256);
    const dim3 grd(D / 128, M / 128);
    const dim3 tgrd(D / 32, D / 32);

    // one-time bf16 conversion / weight pre-transposes
    cvt_bf16_kernel<<<dim3((unsigned)(((long long)M * D / 4 + 255) / 256)),
                      blk, 0, stream>>>(mu, mub, (long long)M * D / 4);
    const size_t wslot = (size_t)D * D;
    transpose_bf16_kernel<<<tgrd, blk, 0, stream>>>(Wq,              Wt + 0 * wslot, D, D, D);
    transpose_bf16_kernel<<<tgrd, blk, 0, stream>>>(Wk,              Wt + 1 * wslot, D, D, D);
    transpose_bf16_kernel<<<tgrd, blk, 0, stream>>>(Wv,              Wt + 2 * wslot, D, D, D);
    transpose_bf16_kernel<<<tgrd, blk, 0, stream>>>(Wm1,             Wt + 3 * wslot, D, D, D);
    transpose_bf16_kernel<<<tgrd, blk, 0, stream>>>(Wm1 + wslot,     Wt + 4 * wslot, D, D, D);
    transpose_bf16_kernel<<<tgrd, blk, 0, stream>>>(Wm2,             Wt + 5 * wslot, D, D, D);
    transpose_bf16_kernel<<<tgrd, blk, 0, stream>>>(Wo,              Wt + 6 * wslot, D, D, D);
    transpose_bf16_kernel<<<tgrd, blk, 0, stream>>>(Wo + wslot,      Wt + 7 * wslot, D, D, D);

    // q,k (bf16), v (f32), A = mu@Wm1_top + bm1, Bm = mu@Wm1_bot
    gemm_wmma_bf16<true,  true,  false><<<grd, blk, 0, stream>>>(mub, Wt + 0 * wslot, bq,      qb,   M, D, D);
    gemm_wmma_bf16<true,  true,  false><<<grd, blk, 0, stream>>>(mub, Wt + 1 * wslot, bk,      kbuf, M, D, D);
    gemm_wmma_bf16<false, true,  false><<<grd, blk, 0, stream>>>(mub, Wt + 2 * wslot, bv,      vbuf, M, D, D);
    gemm_wmma_bf16<false, true,  false><<<grd, blk, 0, stream>>>(mub, Wt + 3 * wslot, bm1,     Abuf, M, D, D);
    gemm_wmma_bf16<false, false, false><<<grd, blk, 0, stream>>>(mub, Wt + 4 * wslot, nullptr, Bbuf, M, D, D);
    // s = mean_w silu(A + shift(Bm))  (bf16, overlays mu-bf16 which is dead)
    silu_window_mean<<<dim3((unsigned)((long long)M * D / 256)), blk, 0,
                      stream>>>(Abuf, Bbuf, sb, N, D);
    // local = s @ Wm2 + bm2  (bf16, overlays A which is dead)
    gemm_wmma_bf16<true,  true,  false><<<grd, blk, 0, stream>>>(sb, Wt + 5 * wslot, bm2, localb, M, D, D);
    // attention -> glob (bf16, overlays s which is dead)
    attn_topk_kernel<<<dim3(M / 16), dim3(128), 0, stream>>>(qb, kbuf, vbuf,
                                                             globb, N, D, scale);
    // out = local @ Wo_top + bo ; out += glob @ Wo_bot
    gemm_wmma_bf16<false, true,  false><<<grd, blk, 0, stream>>>(localb, Wt + 6 * wslot, bo,      d_out, M, D, D);
    gemm_wmma_bf16<false, false, true ><<<grd, blk, 0, stream>>>(globb,  Wt + 7 * wslot, nullptr, d_out, M, D, D);
}
